// RNNTScanLoss_34437047779353
// MI455X (gfx1250) — matrix operand order repaired and checked by reading, not compiled, not run on hardware
//
#include <hip/hip_runtime.h>
#include <hip/hip_bf16.h>
#include <stddef.h>

#define LOG0f (-1e30f)

typedef __attribute__((ext_vector_type(16))) __bf16 v16bf;
typedef __attribute__((ext_vector_type(8)))  __bf16 v8bf;
typedef __attribute__((ext_vector_type(8)))  float  v8f;

union FragBF {
  v16bf v;
  v8bf  h[2];
};

// -------- Kernel 0: W_out (J=512 x V=1024) f32 -> W_outT (V=1024 x J=512) bf16
__global__ __launch_bounds__(256) void k_transpose_wout(
    const float* __restrict__ W_out, __bf16* __restrict__ WoT)
{
  int idx = blockIdx.x * 256 + threadIdx.x;   // 0 .. 512*1024-1
  int v = idx >> 9;                           // 0..1023
  int j = idx & 511;                          // 0..511
  WoT[idx] = (__bf16)W_out[(size_t)j * 1024 + v];
}

// -------- Kernel 1: Y(rows x 512) = X(rows x K) @ W(K x 512), f32 (small GEMMs)
__global__ __launch_bounds__(256) void k_proj(
    const float* __restrict__ X, const float* __restrict__ W,
    float* __restrict__ Y, int K)
{
  int tid = threadIdx.x;
  int row = blockIdx.x * 2 + (tid >> 7);
  int col = blockIdx.y * 128 + (tid & 127);
  const float* x = X + (size_t)row * K;
  const float* w = W + col;
  float acc = 0.f;
#pragma unroll 8
  for (int k = 0; k < K; ++k)
    acc = fmaf(x[k], w[(size_t)k * 512], acc);
  Y[(size_t)row * 512 + col] = acc;
}

__device__ __forceinline__ float grp16_max(float v) {
  v = fmaxf(v, __shfl_xor(v, 1, 32));
  v = fmaxf(v, __shfl_xor(v, 2, 32));
  v = fmaxf(v, __shfl_xor(v, 4, 32));
  v = fmaxf(v, __shfl_xor(v, 8, 32));
  return v;
}
__device__ __forceinline__ float grp16_sum(float v) {
  v += __shfl_xor(v, 1, 32);
  v += __shfl_xor(v, 2, 32);
  v += __shfl_xor(v, 4, 32);
  v += __shfl_xor(v, 8, 32);
  return v;
}

// -------- Kernel 2: fused joint network per (b,t):
//   h = tanh(ep + pp + b_joint) -> bf16 LDS (80x512, rows 65..79 zero pad)
//   logits = h @ W_outT (bf16 WMMA, f32 acc) + b_out, streamed logsumexp over V
//   emit blank_lp[b,t,u] and label_lp[b,t,u]
__global__ __launch_bounds__(320) void k_joint(
    const float* __restrict__ ep, const float* __restrict__ pp,
    const float* __restrict__ bj, const __bf16* __restrict__ WoT,
    const float* __restrict__ bout, const int* __restrict__ labels,
    const int* __restrict__ blank_ptr,
    float* __restrict__ blankLp, float* __restrict__ labelLp)
{
  const int HROW = 520;                       // 1040B pitch: 4-bank rotation/row
  __shared__ __bf16 hbuf[80 * 520];           // 83,200 B
  __shared__ float eps[512];
  __shared__ float bjs[512];
  __shared__ float part_m[2][80];
  __shared__ float part_s[2][80];
  __shared__ float blankLog[80];
  __shared__ float labLog[80];
  __shared__ int   labs[64];

  int tid = threadIdx.x;
  int bt = blockIdx.x;                        // b*256 + t
  int b = bt >> 8;

  for (int j = tid; j < 512; j += 320) {
    eps[j] = ep[(size_t)bt * 512 + j];
    bjs[j] = bj[j];
  }
  if (tid < 64) labs[tid] = labels[b * 64 + tid];
  __syncthreads();

  const float* ppb = pp + (size_t)b * 65 * 512;
  for (int idx = tid; idx < 80 * 512; idx += 320) {
    int u = idx >> 9;
    int j = idx & 511;
    float v = 0.f;
    if (u < 65) v = tanhf(eps[j] + ppb[(size_t)u * 512 + j] + bjs[j]);
    hbuf[u * HROW + j] = (__bf16)v;
  }
  __syncthreads();

  int lane = tid & 31;
  int wv   = tid >> 5;                        // 0..9
  int mt   = wv >> 1;                         // m-tile 0..4 (rows mt*16..+15)
  int half = wv & 1;                          // n-tile half: 0..31 / 32..63
  int lrow = lane & 15;
  int lhi  = lane >> 4;
  int blankv = blank_ptr[0];

  // Cache all 16 A fragments (ISA 16-bit A layout: lane=row, chunks at lhi*8, +16)
  FragBF afrag[16];
  const __bf16* arow = &hbuf[(mt * 16 + lrow) * HROW];
#pragma unroll
  for (int kk = 0; kk < 16; ++kk) {
    int base = kk * 32 + lhi * 8;
    afrag[kk].h[0] = *(const v8bf*)(arow + base);
    afrag[kk].h[1] = *(const v8bf*)(arow + base + 16);
  }

  float run_m[8], run_s[8];
#pragma unroll
  for (int r = 0; r < 8; ++r) { run_m[r] = LOG0f; run_s[r] = 0.f; }

  for (int i = 0; i < 32; ++i) {
    int nt = half * 32 + i;
    // B layout: lane=column, chunks at lhi*8, +16 (same K order as A)
    const __bf16* brow = WoT + (size_t)(nt * 16 + lrow) * 512 + lhi * 8;
    v8f acc = {0.f, 0.f, 0.f, 0.f, 0.f, 0.f, 0.f, 0.f};
#pragma unroll
    for (int kk = 0; kk < 16; ++kk) {
      FragBF bfr;
      bfr.h[0] = *(const v8bf*)(brow + kk * 32);
      bfr.h[1] = *(const v8bf*)(brow + kk * 32 + 16);
      acc = __builtin_amdgcn_wmma_f32_16x16x32_bf16(
          false, afrag[kk].v, false, bfr.v, (short)0, acc, false, false);
    }
    float bval = bout[nt * 16 + lrow];
#pragma unroll
    for (int r = 0; r < 8; ++r) {
      float x = acc[r] + bval;                // logit, 16 cols across lane group
      float m = grp16_max(x);
      float s = grp16_sum(__expf(x - m));
      float nm = fmaxf(run_m[r], m);
      run_s[r] = run_s[r] * __expf(run_m[r] - nm) + s * __expf(m - nm);
      run_m[r] = nm;
      int row = mt * 16 + r + (lhi << 3);     // C layout: VGPR r -> rows r, r+8
      int col = (nt << 4) + lrow;
      if (col == blankv && row < 65) blankLog[row] = x;
      if (row < 64 && col == labs[row]) labLog[row] = x;
    }
  }

  if (lrow == 0) {
#pragma unroll
    for (int r = 0; r < 8; ++r) {
      int row = mt * 16 + r + (lhi << 3);
      part_m[half][row] = run_m[r];
      part_s[half][row] = run_s[r];
    }
  }
  __syncthreads();

  if (tid < 65) {
    int u = tid;
    float m0 = part_m[0][u], s0 = part_s[0][u];
    float m1 = part_m[1][u], s1 = part_s[1][u];
    float M = fmaxf(m0, m1);
    float lse = M + __logf(s0 * __expf(m0 - M) + s1 * __expf(m1 - M));
    blankLp[(size_t)bt * 65 + u] = blankLog[u] - lse;
    if (u < 64) labelLp[(size_t)bt * 64 + u] = labLog[u] - lse;
  }
}

// -------- Kernel 3: alpha scan over T (cumsum + prefix-logsumexp per batch)
__global__ __launch_bounds__(1024) void k_scan(
    const float* __restrict__ blankLp, const float* __restrict__ labelLp,
    const int* __restrict__ enc_lens, const int* __restrict__ label_lens,
    float* __restrict__ out)
{
  const int T = 256, Up1 = 65, U = 64;
  int tid = threadIdx.x;
  int b = tid >> 7;
  int u = tid & 127;
  __shared__ float alpha[8][66];
  __shared__ float sc[2][8][132];
  __shared__ float contrib[8];
  if (u < Up1) alpha[b][u] = (u == 0) ? 0.f : LOG0f;
  __syncthreads();
  int el = enc_lens[b];
  for (int t = 0; t < T; ++t) {
    float blp = (u < Up1) ? blankLp[(size_t)(b * T + t) * Up1 + u] : LOG0f;
    float llp = (u < U) ? labelLp[(size_t)(b * T + t) * U + u] : 0.f;
    // inclusive cumsum of llp
    sc[0][b][u] = llp;
    __syncthreads();
    int p = 0;
    for (int d = 1; d < 128; d <<= 1) {
      float v = sc[p][b][u];
      if (u >= d) v += sc[p][b][u - d];
      sc[1 - p][b][u] = v;
      __syncthreads();
      p ^= 1;
    }
    float Lval = (u == 0) ? 0.f : ((u <= U) ? sc[p][b][u - 1] : 0.f);
    float y = (u < Up1) ? (alpha[b][u] + blp - Lval) : LOG0f;
    __syncthreads();
    // inclusive prefix logsumexp of y
    sc[0][b][u] = y;
    __syncthreads();
    p = 0;
    for (int d = 1; d < 128; d <<= 1) {
      float a = sc[p][b][u];
      float v = a;
      if (u >= d) {
        float c = sc[p][b][u - d];
        float m = fmaxf(a, c);
        v = m + __logf(__expf(a - m) + __expf(c - m));
      }
      sc[1 - p][b][u] = v;
      __syncthreads();
      p ^= 1;
    }
    float anew = Lval + sc[p][b][u];
    __syncthreads();
    if (u < Up1 && t < el) alpha[b][u] = anew;
    __syncthreads();
  }
  if (u == 0) {
    int ll = label_lens[b];
    float tb = blankLp[(size_t)(b * T + (el - 1)) * Up1 + ll];
    contrib[b] = -(alpha[b][ll] + tb);
  }
  __syncthreads();
  if (tid == 0) {
    float s = 0.f;
    for (int i = 0; i < 8; ++i) s += contrib[i];
    out[0] = s * 0.125f;
  }
}

extern "C" void kernel_launch(void* const* d_in, const int* in_sizes, int n_in,
                              void* d_out, int out_size, void* d_ws, size_t ws_size,
                              hipStream_t stream) {
  const float* enc        = (const float*)d_in[0];
  const float* pred       = (const float*)d_in[1];
  const float* W_enc      = (const float*)d_in[2];
  const float* W_pred     = (const float*)d_in[3];
  const float* b_joint    = (const float*)d_in[4];
  const float* W_out      = (const float*)d_in[5];
  const float* b_out      = (const float*)d_in[6];
  const int*   labels     = (const int*)d_in[7];
  const int*   enc_lens   = (const int*)d_in[8];
  const int*   label_lens = (const int*)d_in[9];
  const int*   blank_id   = (const int*)d_in[10];
  float* out = (float*)d_out;

  // workspace carve-up (all fully rewritten each call; ~7.0 MB total)
  char* w = (char*)d_ws;
  float*  ep      = (float*)(w);                                       // 2048*512*4   = 4,194,304
  float*  pp      = (float*)(w + 4194304);                             //  520*512*4   = 1,064,960
  __bf16* WoT     = (__bf16*)(w + 4194304 + 1064960);                  // 1024*512*2   = 1,048,576
  float*  blankLp = (float*)(w + 4194304 + 1064960 + 1048576);         // 2048*65*4    =   532,480
  float*  labelLp = (float*)(w + 4194304 + 1064960 + 1048576 + 532480);// 2048*64*4    =   524,288

  k_transpose_wout<<<2048, 256, 0, stream>>>(W_out, WoT);
  k_proj<<<dim3(1024, 4), 256, 0, stream>>>(enc, W_enc, ep, 640);      // ep: 2048 rows
  k_proj<<<dim3(260, 4), 256, 0, stream>>>(pred, W_pred, pp, 640);     // pp: 520 rows
  k_joint<<<2048, 320, 0, stream>>>(ep, pp, b_joint, WoT, b_out, labels,
                                    blank_id, blankLp, labelLp);
  k_scan<<<1, 1024, 0, stream>>>(blankLp, labelLp, enc_lens, label_lens, out);

  (void)in_sizes; (void)n_in; (void)out_size; (void)ws_size;
}
